// NeuronSAT_34746285425115
// MI455X (gfx1250) — matrix operand order repaired and checked by reading, not compiled, not run on hardware
//
#include <hip/hip_runtime.h>
#include <hip/hip_bf16.h>
#include <math.h>

typedef __attribute__((ext_vector_type(16))) _Float16 v16h;
typedef __attribute__((ext_vector_type(8)))  _Float16 v8h;
typedef __attribute__((ext_vector_type(8)))  float    v8f;

#define NG    8
#define NV    400
#define NC    1200
#define KLIT  5
#define NNG   2000        // nodes per graph
#define NTOT  16000       // total nodes
#define D     128
#define LROWS 6400        // total literal nodes (NG * 2NV)
#define CROWS 9600        // total clause nodes (NG * NC)
#define NEDGE 48000       // NG*NC*KLIT
#define ROUNDS 26

#define CDIV(a,b) (((a)+(b)-1)/(b))

// ---------------------------------------------------------------------------
// WMMA GEMM: Y[M x Nn] = act( A[M x Kd](f16) * W[Nn x Kd]^T(f16) + bias )
//   optional accumulate into existing f32 Y, optional relu,
//   output either f32 (Yf) or f16 (Yh).
// Block = 256 threads (8 wave32). Block tile = 32 rows x 128 cols; each wave
// owns a 32x16 output slab (two 16x16 accumulators) and loops K in steps of
// 32 with v_wmma_f32_16x16x32_f16. Each B fragment (the only global traffic
// in the k-loop) is reused by BOTH accumulators -> half the B bytes per FLOP.
// A tile staged in LDS (32 x Kd halves, <=16KB).
// ---------------------------------------------------------------------------
__global__ __launch_bounds__(256) void gemm_wmma_kernel(
    const _Float16* __restrict__ A, const _Float16* __restrict__ W,
    const float* __restrict__ bias, float* __restrict__ Yf,
    _Float16* __restrict__ Yh, int M, int Kd, int Nn,
    int accumulate, int relu)
{
    __shared__ _Float16 As[32 * 256];   // Kd <= 256

    const int m0   = blockIdx.x * 32;
    const int nblk = blockIdx.y * 128;
    const int tid  = threadIdx.x;
    const int wave = tid >> 5;
    const int lane = tid & 31;
    const int Kd8  = Kd >> 3;

    // cooperative A-tile load (32 rows x Kd halves), 16B vectors
    for (int i = tid; i < 4 * Kd; i += 256) {   // 32*Kd/8 == 4*Kd chunks
        int row  = i / Kd8;
        int c8   = i - row * Kd8;
        int grow = m0 + row;
        v8h v;
        if (grow < M) {
            v = *(const v8h*)(A + (size_t)grow * Kd + (size_t)c8 * 8);
        } else {
#pragma unroll
            for (int j = 0; j < 8; j++) v[j] = (_Float16)0.0f;
        }
        *(v8h*)(&As[row * Kd + c8 * 8]) = v;
    }
    __syncthreads();

    const int n0 = nblk + wave * 16;
    if (n0 >= Nn) return;

    const int hs   = lane >> 4;        // half-wave select
    const int arow = lane & 15;        // A row (within 16-row tile) this lane carries
    const int ncol = n0 + (lane & 15); // output column this lane carries

    v8f acc0 = {};
    v8f acc1 = {};
    const _Float16* wrow = W + (size_t)ncol * Kd + hs * 8;

    for (int k0 = 0; k0 < Kd; k0 += 32) {
        // B fragment: lane n = ncol, K = {k0+8hs..+7, k0+16+8hs..+7} (matches A)
        v8h blo = *(const v8h*)(wrow + k0);
        v8h bhi = *(const v8h*)(wrow + k0 + 16);
        v16h b;
#pragma unroll
        for (int j = 0; j < 8; j++) { b[j] = blo[j]; b[j + 8] = bhi[j]; }
        __builtin_prefetch(wrow + k0 + 64, 0, 1);   // global_prefetch_b8 next chunk

        // A fragments per ISA 16-bit 16x32 layout:
        //   lanes 0-15: K = {k0..k0+7, k0+16..k0+23}
        //   lanes16-31: K = {k0+8..k0+15, k0+24..k0+31}
        v8h a0lo = *(const v8h*)(&As[arow * Kd + k0 + hs * 8]);
        v8h a0hi = *(const v8h*)(&As[arow * Kd + k0 + 16 + hs * 8]);
        v8h a1lo = *(const v8h*)(&As[(arow + 16) * Kd + k0 + hs * 8]);
        v8h a1hi = *(const v8h*)(&As[(arow + 16) * Kd + k0 + 16 + hs * 8]);
        v16h a0, a1;
#pragma unroll
        for (int j = 0; j < 8; j++) {
            a0[j] = a0lo[j]; a0[j + 8] = a0hi[j];
            a1[j] = a1lo[j]; a1[j + 8] = a1hi[j];
        }

        acc0 = __builtin_amdgcn_wmma_f32_16x16x32_f16(
            false, a0, false, b, (short)0, acc0, false, false);
        acc1 = __builtin_amdgcn_wmma_f32_16x16x32_f16(
            false, a1, false, b, (short)0, acc1, false, false);
    }

    // C/D layout: VGPR r -> M = r (lanes 0-15) or r+8 (lanes 16-31); N = lane&15
#pragma unroll
    for (int r = 0; r < 8; r++) {
        int grow0 = m0 + r + hs * 8;
        int grow1 = grow0 + 16;
        float val0 = acc0[r];
        float val1 = acc1[r];
        if (bias) { float bv = bias[ncol]; val0 += bv; val1 += bv; }
        if (grow0 < M) {
            if (accumulate && Yf) val0 += Yf[(size_t)grow0 * Nn + ncol];
            if (relu) val0 = fmaxf(val0, 0.0f);
            if (Yf) Yf[(size_t)grow0 * Nn + ncol] = val0;
            if (Yh) Yh[(size_t)grow0 * Nn + ncol] = (_Float16)val0;
        }
        if (grow1 < M) {
            if (accumulate && Yf) val1 += Yf[(size_t)grow1 * Nn + ncol];
            if (relu) val1 = fmaxf(val1, 0.0f);
            if (Yf) Yf[(size_t)grow1 * Nn + ncol] = val1;
            if (Yh) Yh[(size_t)grow1 * Nn + ncol] = (_Float16)val1;
        }
    }
}

// ---------------------------------------------------------------------------
// Elementwise / gather / scatter helpers
// ---------------------------------------------------------------------------
__global__ void cvt_f32_to_f16(const float* __restrict__ s, _Float16* __restrict__ d, int n) {
    int i = blockIdx.x * blockDim.x + threadIdx.x;
    if (i < n) d[i] = (_Float16)s[i];
}

__global__ void pad_vote_w2(const float* __restrict__ w2, _Float16* __restrict__ dst) {
    int i = blockIdx.x * blockDim.x + threadIdx.x;      // 16*128
    if (i >= 16 * D) return;
    int row = i >> 7, d = i & 127;
    dst[i] = (_Float16)(row == 0 ? w2[d] : 0.0f);
}

__global__ void init_hc(const float* __restrict__ Lw, const float* __restrict__ Lb,
                        const float* __restrict__ Cw, const float* __restrict__ Cb,
                        float* __restrict__ h, float* __restrict__ c) {
    int i = blockIdx.x * blockDim.x + threadIdx.x;
    if (i >= NTOT * D) return;
    int n = i >> 7, d = i & 127;
    bool lit = (n % NNG) < 2 * NV;
    h[i] = lit ? (Lw[d] + Lb[d]) : (Cw[d] + Cb[d]);
    c[i] = 0.0f;
}

__global__ void gather_cvt(const float* __restrict__ h, const int* __restrict__ idx,
                           _Float16* __restrict__ out, int rows) {
    int i = blockIdx.x * blockDim.x + threadIdx.x;
    if (i >= rows * D) return;
    int r = i >> 7, d = i & 127;
    out[i] = (_Float16)h[(size_t)idx[r] * D + d];
}

// clause j gathers its 5 contiguous edges (edge layout is (g, clause, k)-major)
__global__ void agg_clause(const _Float16* __restrict__ m, const int* __restrict__ edge_src,
                           _Float16* __restrict__ aggc) {
    int i = blockIdx.x * blockDim.x + threadIdx.x;
    if (i >= CROWS * D) return;
    int cj = i >> 7, d = i & 127;
    int base = cj * KLIT;
    float s = 0.0f;
#pragma unroll
    for (int k = 0; k < KLIT; k++) {
        int src = edge_src[base + k];            // literal node id
        int lit = (src / NNG) * (2 * NV) + (src % NNG);
        s += (float)m[(size_t)lit * D + d];
    }
    aggc[i] = (_Float16)s;
}

__global__ void zero_f32(float* __restrict__ p, int n) {
    int i = blockIdx.x * blockDim.x + threadIdx.x;
    if (i < n) p[i] = 0.0f;
}

__global__ void scatter_lit(const _Float16* __restrict__ m2, const int* __restrict__ edge_src,
                            const int* __restrict__ edge_dst, float* __restrict__ aggl) {
    int i = blockIdx.x * blockDim.x + threadIdx.x;
    if (i >= NEDGE * D) return;
    int e = i >> 7, d = i & 127;
    int src = edge_src[e];
    int dst = edge_dst[e];
    int lit = (src / NNG) * (2 * NV) + (src % NNG);
    int cls = (dst / NNG) * NC + ((dst % NNG) - 2 * NV);
    atomicAdd(&aggl[(size_t)lit * D + d], (float)m2[(size_t)cls * D + d]);
}

// xin = concat( agg_l , h[flip] ) converted to f16, row stride 256
__global__ void pack_xin(const float* __restrict__ aggl, const float* __restrict__ h,
                         const int* __restrict__ flip_index, _Float16* __restrict__ xin) {
    int i = blockIdx.x * blockDim.x + threadIdx.x;
    if (i >= LROWS * D) return;
    int r = i >> 7, d = i & 127;
    xin[(size_t)r * 256 + d]       = (_Float16)aggl[i];
    xin[(size_t)r * 256 + 128 + d] = (_Float16)h[(size_t)flip_index[r] * D + d];
}

__device__ __forceinline__ float sigm(float x) { return 1.0f / (1.0f + __expf(-x)); }

__global__ void lstm_pointwise(const float* __restrict__ gbuf, const float* __restrict__ bih,
                               const float* __restrict__ bhh, const int* __restrict__ node_idx,
                               float* __restrict__ h, float* __restrict__ cst, int rows) {
    int i = blockIdx.x * blockDim.x + threadIdx.x;
    if (i >= rows * D) return;
    int r = i >> 7, d = i & 127;
    const float* g = gbuf + (size_t)r * 512;
    float gi = g[d]       + bih[d]       + bhh[d];
    float gf = g[128 + d] + bih[128 + d] + bhh[128 + d];
    float gg = g[256 + d] + bih[256 + d] + bhh[256 + d];
    float go = g[384 + d] + bih[384 + d] + bhh[384 + d];
    size_t off = (size_t)node_idx[r] * D + d;
    float c2 = sigm(gf) * cst[off] + sigm(gi) * tanhf(gg);
    h[off]   = sigm(go) * tanhf(c2);
    cst[off] = c2;
}

__global__ void vote_reduce(const float* __restrict__ vfull, const float* __restrict__ vb2,
                            const int* __restrict__ n_vars, float* __restrict__ out) {
    __shared__ float red[256];
    int g = blockIdx.x;
    float s = 0.0f;
    for (int r = threadIdx.x; r < 2 * NV; r += 256)
        s += vfull[(size_t)(g * 2 * NV + r) * 16] + vb2[0];
    red[threadIdx.x] = s;
    __syncthreads();
    for (int st = 128; st > 0; st >>= 1) {
        if (threadIdx.x < st) red[threadIdx.x] += red[threadIdx.x + st];
        __syncthreads();
    }
    if (threadIdx.x == 0) out[g] = red[0] / (2.0f * (float)n_vars[g]);
}

// ---------------------------------------------------------------------------
extern "C" void kernel_launch(void* const* d_in, const int* in_sizes, int n_in,
                              void* d_out, int out_size, void* d_ws, size_t ws_size,
                              hipStream_t stream) {
    (void)in_sizes; (void)n_in; (void)out_size; (void)ws_size;

    const int* edge_src   = (const int*)d_in[0];
    const int* edge_dst   = (const int*)d_in[1];
    const int* l_index    = (const int*)d_in[2];
    const int* c_index    = (const int*)d_in[3];
    const int* flip_index = (const int*)d_in[4];
    /* batch = d_in[5] unused */
    const int* n_vars     = (const int*)d_in[6];
    const float* L_init_w = (const float*)d_in[7];
    const float* L_init_b = (const float*)d_in[8];
    const float* C_init_w = (const float*)d_in[9];
    const float* C_init_b = (const float*)d_in[10];
    const float* lmsg_w   = (const float*)d_in[11];
    const float* lmsg_b   = (const float*)d_in[12];
    const float* cmsg_w   = (const float*)d_in[13];
    const float* cmsg_b   = (const float*)d_in[14];
    const float* lu_wih   = (const float*)d_in[15];
    const float* lu_whh   = (const float*)d_in[16];
    const float* lu_bih   = (const float*)d_in[17];
    const float* lu_bhh   = (const float*)d_in[18];
    const float* cu_wih   = (const float*)d_in[19];
    const float* cu_whh   = (const float*)d_in[20];
    const float* cu_bih   = (const float*)d_in[21];
    const float* cu_bhh   = (const float*)d_in[22];
    const float* vote_w0  = (const float*)d_in[23];
    const float* vote_b0  = (const float*)d_in[24];
    const float* vote_w1  = (const float*)d_in[25];
    const float* vote_b1  = (const float*)d_in[26];
    const float* vote_w2  = (const float*)d_in[27];
    const float* vote_b2  = (const float*)d_in[28];
    float* out = (float*)d_out;

    // -------- workspace carve --------
    char* p = (char*)d_ws;
    auto carve = [&](size_t bytes) -> char* {
        char* r = p;
        p += (bytes + 255) & ~(size_t)255;
        return r;
    };
    float*     h        = (float*)    carve((size_t)NTOT * D * 4);
    float*     cst      = (float*)    carve((size_t)NTOT * D * 4);
    _Float16*  w_lmsg   = (_Float16*) carve((size_t)3 * D * D * 2);
    _Float16*  w_cmsg   = (_Float16*) carve((size_t)3 * D * D * 2);
    _Float16*  w_cu_wih = (_Float16*) carve((size_t)512 * 128 * 2);
    _Float16*  w_cu_whh = (_Float16*) carve((size_t)512 * 128 * 2);
    _Float16*  w_lu_wih = (_Float16*) carve((size_t)512 * 256 * 2);
    _Float16*  w_lu_whh = (_Float16*) carve((size_t)512 * 128 * 2);
    _Float16*  w_v0     = (_Float16*) carve((size_t)D * D * 2);
    _Float16*  w_v1     = (_Float16*) carve((size_t)D * D * 2);
    _Float16*  w_v2p    = (_Float16*) carve((size_t)16 * D * 2);
    _Float16*  xin      = (_Float16*) carve((size_t)LROWS * 256 * 2);
    _Float16*  hgat     = (_Float16*) carve((size_t)CROWS * D * 2);
    _Float16*  t1       = (_Float16*) carve((size_t)CROWS * D * 2);
    _Float16*  t2       = (_Float16*) carve((size_t)CROWS * D * 2);
    _Float16*  mbuf     = (_Float16*) carve((size_t)CROWS * D * 2);
    _Float16*  aggc     = (_Float16*) carve((size_t)CROWS * D * 2);
    float*     aggl     = (float*)    carve((size_t)LROWS * D * 4);
    float*     gbuf     = (float*)    carve((size_t)CROWS * 512 * 4);
    float*     vfull    = (float*)    carve((size_t)LROWS * 16 * 4);

    auto cvt = [&](const float* s, _Float16* d, int n) {
        cvt_f32_to_f16<<<CDIV(n, 256), 256, 0, stream>>>(s, d, n);
    };
    auto gemm = [&](const _Float16* A, const _Float16* W, const float* bias,
                    float* Yf, _Float16* Yh, int M, int Kd, int Nn, int acc, int relu) {
        dim3 grid(CDIV(M, 32), CDIV(Nn, 128));
        gemm_wmma_kernel<<<grid, 256, 0, stream>>>(A, W, bias, Yf, Yh, M, Kd, Nn, acc, relu);
    };

    // -------- one-time f16 weight packing (re-done each launch: deterministic) --------
    cvt(lmsg_w, w_lmsg, 3 * D * D);
    cvt(cmsg_w, w_cmsg, 3 * D * D);
    cvt(cu_wih, w_cu_wih, 512 * 128);
    cvt(cu_whh, w_cu_whh, 512 * 128);
    cvt(lu_wih, w_lu_wih, 512 * 256);
    cvt(lu_whh, w_lu_whh, 512 * 128);
    cvt(vote_w0, w_v0, D * D);
    cvt(vote_w1, w_v1, D * D);
    pad_vote_w2<<<CDIV(16 * D, 256), 256, 0, stream>>>(vote_w2, w_v2p);

    init_hc<<<CDIV(NTOT * D, 256), 256, 0, stream>>>(L_init_w, L_init_b, C_init_w, C_init_b, h, cst);

    // -------- message-passing rounds --------
    for (int rd = 0; rd < ROUNDS; rd++) {
        // literal messages: m = MLP3(h[lits])
        gather_cvt<<<CDIV(LROWS * D, 256), 256, 0, stream>>>(h, l_index, hgat, LROWS);
        gemm(hgat, w_lmsg,             lmsg_b,         nullptr, t1,   LROWS, D, D, 0, 1);
        gemm(t1,   w_lmsg + D * D,     lmsg_b + D,     nullptr, t2,   LROWS, D, D, 0, 1);
        gemm(t2,   w_lmsg + 2 * D * D, lmsg_b + 2 * D, nullptr, mbuf, LROWS, D, D, 0, 0);

        // clause aggregation (regular 5-edge gather) + clause LSTM
        agg_clause<<<CDIV(CROWS * D, 256), 256, 0, stream>>>(mbuf, edge_src, aggc);
        gather_cvt<<<CDIV(CROWS * D, 256), 256, 0, stream>>>(h, c_index, hgat, CROWS);
        gemm(aggc, w_cu_wih, nullptr, gbuf, nullptr, CROWS, D, 512, 0, 0);
        gemm(hgat, w_cu_whh, nullptr, gbuf, nullptr, CROWS, D, 512, 1, 0);
        lstm_pointwise<<<CDIV(CROWS * D, 256), 256, 0, stream>>>(gbuf, cu_bih, cu_bhh,
                                                                 c_index, h, cst, CROWS);

        // clause messages: m2 = MLP3(h[clauses])  (clause h now updated)
        gather_cvt<<<CDIV(CROWS * D, 256), 256, 0, stream>>>(h, c_index, hgat, CROWS);
        gemm(hgat, w_cmsg,             cmsg_b,         nullptr, t1,   CROWS, D, D, 0, 1);
        gemm(t1,   w_cmsg + D * D,     cmsg_b + D,     nullptr, t2,   CROWS, D, D, 0, 1);
        gemm(t2,   w_cmsg + 2 * D * D, cmsg_b + 2 * D, nullptr, mbuf, CROWS, D, D, 0, 0);

        // literal aggregation (irregular -> f32 atomics) + literal LSTM
        zero_f32<<<CDIV(LROWS * D, 256), 256, 0, stream>>>(aggl, LROWS * D);
        scatter_lit<<<CDIV(NEDGE * D, 256), 256, 0, stream>>>(mbuf, edge_src, edge_dst, aggl);
        pack_xin<<<CDIV(LROWS * D, 256), 256, 0, stream>>>(aggl, h, flip_index, xin);
        gather_cvt<<<CDIV(LROWS * D, 256), 256, 0, stream>>>(h, l_index, hgat, LROWS);
        gemm(xin,  w_lu_wih, nullptr, gbuf, nullptr, LROWS, 256, 512, 0, 0);
        gemm(hgat, w_lu_whh, nullptr, gbuf, nullptr, LROWS, D,   512, 1, 0);
        lstm_pointwise<<<CDIV(LROWS * D, 256), 256, 0, stream>>>(gbuf, lu_bih, lu_bhh,
                                                                 l_index, h, cst, LROWS);
    }

    // -------- vote head + per-graph mean --------
    gather_cvt<<<CDIV(LROWS * D, 256), 256, 0, stream>>>(h, l_index, hgat, LROWS);
    gemm(hgat, w_v0,  vote_b0, nullptr, t1,    LROWS, D, D,  0, 1);
    gemm(t1,   w_v1,  vote_b1, nullptr, t2,    LROWS, D, D,  0, 1);
    gemm(t2,   w_v2p, nullptr, vfull,   nullptr, LROWS, D, 16, 0, 0);
    vote_reduce<<<NG, 256, 0, stream>>>(vfull, vote_b2, n_vars, out);
}
